// Prestroid_13134009991195
// MI455X (gfx1250) — matrix operand order repaired and checked by reading, not compiled, not run on hardware
//
#include <hip/hip_runtime.h>
#include <hip/hip_bf16.h>

// Problem constants (match reference)
#define B_BATCH 256
#define NNODES  1023
#define NPAD    1024
#define IDX_STRIDE (3 * NNODES)   // 3069 ints per batch

typedef __attribute__((ext_vector_type(16))) __bf16 v16bf;
typedef __attribute__((ext_vector_type(8)))  __bf16 v8bf;
typedef __attribute__((ext_vector_type(8)))  float  v8f;

// ---- workspace layout (bytes). Total ~269 MB. ----
static constexpr size_t OFF_H1    = 0;                                // bf16 [256][256][1024] = 128 MB
static constexpr size_t OFF_H2    = OFF_H1 + (size_t)256*256*1024*2;  // bf16 [256][128][1024] = 64 MB
static constexpr size_t OFF_H3    = OFF_H2 + (size_t)256*128*1024*2;  // f32  [256][ 64][1024] = 64 MB
static constexpr size_t OFF_W1B   = OFF_H3 + (size_t)256*64*1024*4;   // bf16 256*192
static constexpr size_t OFF_W2B   = OFF_W1B + (size_t)256*192*2;      // bf16 128*768
static constexpr size_t OFF_W3B   = OFF_W2B + (size_t)128*768*2;      // bf16 64*384
static constexpr size_t OFF_STATS = OFF_W3B + (size_t)64*384*2;       // 6*256 f32

// ---------- weight f32 -> bf16 ----------
__global__ void cvt_f32_bf16(const float* __restrict__ src, __bf16* __restrict__ dst, int n) {
    int i = blockIdx.x * blockDim.x + threadIdx.x;
    if (i < n) dst[i] = (__bf16)src[i];
}

// ---------- zero node-0 column of all three activation buffers ----------
__global__ void zero_col0(__bf16* __restrict__ h1, __bf16* __restrict__ h2, float* __restrict__ h3) {
    int i = blockIdx.x * blockDim.x + threadIdx.x;
    if (i < 256 * 256)                   h1[(size_t)i * NPAD] = (__bf16)0.f;
    else if (i < 256 * (256 + 128))      h2[(size_t)(i - 256 * 256) * NPAD] = (__bf16)0.f;
    else if (i < 256 * (256 + 128 + 64)) h3[(size_t)(i - 256 * (256 + 128)) * NPAD] = 0.f;
}

// ---------- fused (norm+leakyrelu of prev layer) -> gather -> WMMA GEMM -> +bias ----------
// One block = one (batch b, 16-wide node tile). COUT/16 waves, one 16x16 C tile each.
// LDS B-tile stored transposed [n][K3+PAD] so each lane's B fragment is two 16B ds loads.
template<int CIN, int COUT, typename PrevT, typename OutT, bool NORM>
__global__ __launch_bounds__(2 * COUT)
void btconv_wmma(const PrevT* __restrict__ prev,     // [B][CIN][1024]
                 const int*   __restrict__ idxes,    // [B][3*1023]
                 const __bf16* __restrict__ Wb,      // [COUT][3*CIN]  (kidx = c*3+kk == flat W layout)
                 const float* __restrict__ bias,     // [COUT]
                 const float* __restrict__ meanArr,  // [B] (NORM only)
                 const float* __restrict__ invArr,   // [B] (NORM only)
                 OutT* __restrict__ out)             // [B][COUT][1024]
{
    constexpr int K3   = 3 * CIN;
    constexpr int K3P  = K3 + 16;                // pad 16 bf16 (32 B) to break bank alignment
    __shared__ __align__(16) __bf16 G[16 * K3P]; // [n][kidx]
    __shared__ int childIdx[16 * 3];

    const int b     = blockIdx.y;
    const int nTile = blockIdx.x;
    const int tid   = threadIdx.x;

    float mean = 0.f, inv = 1.f;
    if constexpr (NORM) { mean = meanArr[b]; inv = invArr[b]; }

    // stage the 48 child-node indices for this node tile
    if (tid < 48) {
        int n = tid / 3, kk = tid % 3;
        int j = nTile * 16 + n;
        childIdx[tid] = (j < NNODES) ? idxes[(size_t)b * IDX_STRIDE + 3 * j + kk] : 0;
    }
    __syncthreads();

    // cooperative gather into LDS (bf16), applying prev-layer norm + leaky-relu on the fly
    const PrevT* prevB = prev + (size_t)b * CIN * NPAD;
    for (int p = tid; p < K3 * 16; p += 2 * COUT) {
        int n    = p & 15;
        int kidx = p >> 4;
        int j    = nTile * 16 + n;
        float v  = 0.f;
        if (j < NNODES) {
            int kk   = kidx % 3;
            int c    = kidx / 3;
            int node = childIdx[n * 3 + kk];
            v = (float)prevB[(size_t)c * NPAD + node];
            if constexpr (NORM) {
                v = (v - mean) * inv;
                v = v > 0.f ? v : 0.01f * v;   // leaky_relu(0.01)
            }
        }
        G[n * K3P + kidx] = (__bf16)v;
    }
    __syncthreads();

    const int wave = tid >> 5;
    const int lane = tid & 31;
    const int half = lane >> 4;     // which 16-lane half
    const int nn   = lane & 15;
    const int coutBase = wave * 16;

    v8f acc = {};
    // A-fragment row for this lane is (coutBase + nn); B-fragment column is nn.
    const __bf16* Wrow = Wb + (size_t)(coutBase + nn) * K3;
    const __bf16* Gn   = G + nn * K3P;

    for (int ks = 0; ks < K3; ks += 32) {
        // A 16x32 bf16: lane<16 holds K {ks..ks+7, ks+16..ks+23}; lane>=16 holds {ks+8..15, ks+24..31}
        v8bf a0 = *(const v8bf*)(Wrow + ks + half * 8);
        v8bf a1 = *(const v8bf*)(Wrow + ks + 16 + half * 8);
        // B 32x16 bf16: lane<16 holds col nn, K {ks..ks+7, ks+16..ks+23}; lane>=16 holds {ks+8..15, ks+24..31}
        v8bf b0 = *(const v8bf*)(Gn + ks + half * 8);
        v8bf b1 = *(const v8bf*)(Gn + ks + 16 + half * 8);

        v16bf a, bm;
        #pragma unroll
        for (int e = 0; e < 8; ++e) {
            a[e]  = a0[e];  a[e + 8]  = a1[e];
            bm[e] = b0[e];  bm[e + 8] = b1[e];
        }
        acc = __builtin_amdgcn_wmma_f32_16x16x32_bf16(
            /*neg_a=*/false, a, /*neg_b=*/false, bm,
            /*c_mod=*/(short)0, acc, /*reuse_a=*/false, /*reuse_b=*/false);
    }

    // C/D f32 layout: VGPR r -> row (r + 8*half), col nn. Store to node j+1 (node 0 is zero pad).
    int j = nTile * 16 + nn;
    if (j < NNODES) {
        OutT* outB = out + (size_t)b * COUT * NPAD;
        #pragma unroll
        for (int r = 0; r < 8; ++r) {
            int m = coutBase + r + half * 8;
            float val = acc[r] + bias[m];
            outB[(size_t)m * NPAD + (j + 1)] = (OutT)val;
        }
    }
}

// ---------- per-tree mean / 1/(std+1e-5), std with ddof=1, over (COUT, 1024) ----------
template<typename T, int COUT>
__global__ __launch_bounds__(256)
void tree_stats(const T* __restrict__ h, float* __restrict__ meanOut, float* __restrict__ invOut) {
    const int b = blockIdx.x;
    const size_t M = (size_t)COUT * NPAD;
    const T* hb = h + (size_t)b * M;
    double s = 0.0, s2 = 0.0;
    for (size_t i = threadIdx.x; i < M; i += 256) {
        float v = (float)hb[i];
        s  += (double)v;
        s2 += (double)v * (double)v;
    }
    __shared__ double sh[256], sh2[256];
    sh[threadIdx.x] = s; sh2[threadIdx.x] = s2;
    __syncthreads();
    for (int ofs = 128; ofs > 0; ofs >>= 1) {
        if (threadIdx.x < ofs) {
            sh[threadIdx.x]  += sh[threadIdx.x + ofs];
            sh2[threadIdx.x] += sh2[threadIdx.x + ofs];
        }
        __syncthreads();
    }
    if (threadIdx.x == 0) {
        double mean = sh[0] / (double)M;
        double var  = (sh2[0] - sh[0] * sh[0] / (double)M) / (double)(M - 1);
        if (var < 0.0) var = 0.0;
        double stdv = sqrt(var);
        meanOut[b] = (float)mean;
        invOut[b]  = (float)(1.0 / (stdv + 1e-5));
    }
}

// ---------- final: normalize h3, mask, max over nodes. One wave per (b, channel). ----------
__global__ __launch_bounds__(256)
void masked_max(const float* __restrict__ h3, const float* __restrict__ meanArr,
                const float* __restrict__ invArr, const int* __restrict__ masks,
                float* __restrict__ out) {
    const int wave = threadIdx.x >> 5;
    const int lane = threadIdx.x & 31;
    const int gid  = blockIdx.x * (blockDim.x >> 5) + wave;   // 0 .. 256*64-1
    const int b = gid >> 6;
    const int m = gid & 63;
    const float mean = meanArr[b], inv = invArr[b];
    const float* hp = h3 + ((size_t)b * 64 + m) * NPAD;
    const int* mk = masks + (size_t)b * NPAD;
    float mx = -3.402823466e38f;
    for (int n = lane; n < NPAD; n += 32) {
        float v = (hp[n] - mean) * inv;
        v = mk[n] ? v : 0.f;
        mx = fmaxf(mx, v);
    }
    #pragma unroll
    for (int o = 16; o > 0; o >>= 1)
        mx = fmaxf(mx, __shfl_xor(mx, o, 32));
    if (lane == 0) out[gid] = mx;
}

extern "C" void kernel_launch(void* const* d_in, const int* in_sizes, int n_in,
                              void* d_out, int out_size, void* d_ws, size_t ws_size,
                              hipStream_t stream) {
    const float* trees = (const float*)d_in[0];
    const int*   idxes = (const int*)d_in[1];
    const int*   masks = (const int*)d_in[2];
    const float* W1 = (const float*)d_in[3];
    const float* b1 = (const float*)d_in[4];
    const float* W2 = (const float*)d_in[5];
    const float* b2 = (const float*)d_in[6];
    const float* W3 = (const float*)d_in[7];
    const float* b3 = (const float*)d_in[8];
    float* out = (float*)d_out;

    char* ws = (char*)d_ws;
    __bf16* h1  = (__bf16*)(ws + OFF_H1);
    __bf16* h2  = (__bf16*)(ws + OFF_H2);
    float*  h3  = (float*)(ws + OFF_H3);
    __bf16* W1b = (__bf16*)(ws + OFF_W1B);
    __bf16* W2b = (__bf16*)(ws + OFF_W2B);
    __bf16* W3b = (__bf16*)(ws + OFF_W3B);
    float* stats = (float*)(ws + OFF_STATS);
    float* mean1 = stats + 0 * 256, *inv1 = stats + 1 * 256;
    float* mean2 = stats + 2 * 256, *inv2 = stats + 3 * 256;
    float* mean3 = stats + 4 * 256, *inv3 = stats + 5 * 256;

    // weights to bf16 (kidx = c*3+kk matches flat [Cout][Cin][3] layout)
    cvt_f32_bf16<<<(256*192 + 255) / 256, 256, 0, stream>>>(W1, W1b, 256 * 192);
    cvt_f32_bf16<<<(128*768 + 255) / 256, 256, 0, stream>>>(W2, W2b, 128 * 768);
    cvt_f32_bf16<<<( 64*384 + 255) / 256, 256, 0, stream>>>(W3, W3b, 64 * 384);
    zero_col0<<<(256 * 448 + 255) / 256, 256, 0, stream>>>(h1, h2, h3);

    dim3 grid(64, 256);  // (node tiles, batch)

    // layer 1: trees(f32, raw) -> h1(bf16)
    btconv_wmma<64, 256, float, __bf16, false><<<grid, 512, 0, stream>>>(
        trees, idxes, W1b, b1, nullptr, nullptr, h1);
    tree_stats<__bf16, 256><<<256, 256, 0, stream>>>(h1, mean1, inv1);

    // layer 2: norm+lrelu(h1) -> h2(bf16)
    btconv_wmma<256, 128, __bf16, __bf16, true><<<grid, 256, 0, stream>>>(
        h1, idxes, W2b, b2, mean1, inv1, h2);
    tree_stats<__bf16, 128><<<256, 256, 0, stream>>>(h2, mean2, inv2);

    // layer 3: norm+lrelu(h2) -> h3(f32)
    btconv_wmma<128, 64, __bf16, float, true><<<grid, 128, 0, stream>>>(
        h2, idxes, W3b, b3, mean2, inv2, h3);
    tree_stats<float, 64><<<256, 256, 0, stream>>>(h3, mean3, inv3);

    // final norm + mask + max -> out [256][64]
    masked_max<<<(256 * 64) / 8, 256, 0, stream>>>(h3, mean3, inv3, masks, out);
}
